// WindowAttention_37434934952504
// MI455X (gfx1250) — compile-verified, matrix operand order checked by reading, not run on hardware
//
#include <hip/hip_runtime.h>
#include <hip/hip_bf16.h>
#include <stdint.h>

typedef __attribute__((ext_vector_type(16))) __bf16 v16bf;
typedef __attribute__((ext_vector_type(8)))  float  v8f;
typedef unsigned short u16;

#define HEADS 16
#define NTOK  49
#define DIMC  512
#define NPAD  64
#define NWIN  64

union FragU { uint4 q[2]; v16bf v; };

static __device__ __forceinline__ u16 f2bf(float f) {
  union { float f; uint32_t u; } x; x.f = f;
  uint32_t u = x.u;
  uint32_t r = u + 0x7FFFu + ((u >> 16) & 1u);   // round-to-nearest-even
  return (u16)(r >> 16);
}

// Per-lane 16-bit A/B fragment load (A 16x32 layout, B loaded from [N][K] storage
// with identical pattern). p points at row_base + kstep + (lane>>4)*8 elements.
// VGPRs 0-3 = elems [0,8), VGPRs 4-7 = elems [16,24) relative to p.
static __device__ __forceinline__ v16bf ldfrag(const u16* p) {
  FragU f;
  f.q[0] = *(const uint4*)(p);
  f.q[1] = *(const uint4*)(p + 16);
  return f.v;
}

static __device__ __forceinline__ v8f wmma_bf16(v16bf a, v16bf b, v8f c) {
  return __builtin_amdgcn_wmma_f32_16x16x32_bf16(false, a, false, b, (short)0, c, false, false);
}

// ---------------- prep: transpose+convert weights to bf16, gather bias ----------------
__global__ void prep_kernel(const float* __restrict__ qkv_w,   // [512][1536]
                            const float* __restrict__ proj_w,  // [512][512]
                            const float* __restrict__ rel_table,  // [169][16]
                            const int*   __restrict__ rel_index,  // [49*49]
                            u16* __restrict__ qkv_wt,   // [1536][512] bf16
                            u16* __restrict__ proj_wt,  // [512][512] bf16
                            float* __restrict__ bias) { // [16][49][49]
  int i = blockIdx.x * blockDim.x + threadIdx.x;
  const int QN = 1536 * 512, PN = 512 * 512, BN = HEADS * NTOK * NTOK;
  if (i < QN) {
    int c = i >> 9, k = i & 511;
    qkv_wt[i] = f2bf(qkv_w[k * 1536 + c]);
  } else if (i < QN + PN) {
    int j = i - QN;
    int c = j >> 9, k = j & 511;
    proj_wt[j] = f2bf(proj_w[k * 512 + c]);
  } else if (i < QN + PN + BN) {
    int j = i - QN - PN;
    int h = j / (NTOK * NTOK);
    int r = j % (NTOK * NTOK);
    bias[j] = rel_table[rel_index[r] * HEADS + h];
  }
}

// ---------------- fused window attention: one block per window-batch ----------------
__global__ __launch_bounds__(256, 1)
void attn_kernel(const float* __restrict__ x,        // [2048][49][512]
                 const float* __restrict__ mask,     // [64][49][49]
                 const float* __restrict__ qkv_b,    // [1536]
                 const float* __restrict__ proj_b,   // [512]
                 const u16*   __restrict__ qkv_wt,   // [1536][512] bf16
                 const u16*   __restrict__ proj_wt,  // [512][512] bf16
                 const float* __restrict__ bias,     // [16][49][49]
                 float* __restrict__ out) {          // [2048][49][512]
  // LDS layout (u16 units): q 32K | k 32K | v^T 32K | x/o union 32K | P 8K  = 272 KB
  __shared__ __align__(16) u16 smem[139264];
  const int QS = 0, KS = 32768, VT = 65536, XO = 98304, PB = 131072;

  const int b    = blockIdx.x;
  const int tid  = threadIdx.x;
  const int wave = tid >> 5;
  const int lane = tid & 31;
  const int lx   = lane & 15;       // column / row-in-tile index
  const int kh   = lane >> 4;       // K-half select for 16-bit fragments

  // ---- Phase 0: stage x (padded to 64 rows, zeros beyond 49) as bf16 ----
  for (int i = tid; i < NPAD * DIMC; i += 256) {
    int n = i >> 9, c = i & 511;
    float v = (n < NTOK) ? x[((size_t)b * NTOK + n) * DIMC + c] : 0.f;
    smem[XO + i] = f2bf(v);
  }
  __syncthreads();

  const float scale = 0.17677669529663687f;   // 32^-0.5

  // ---- Phase 1: QKV = x @ W_qkv + b ; scatter to q/k/v^T LDS (bf16) ----
  for (int i = 0; i < 12; ++i) {
    int colTile = wave * 12 + i;          // 96 column tiles of 16 over 1536
    int col0 = colTile << 4;
    v8f acc[4] = {};
    for (int ks = 0; ks < DIMC; ks += 32) {
      v16bf bf = ldfrag(qkv_wt + (size_t)(col0 + lx) * DIMC + ks + kh * 8);
#pragma unroll
      for (int r = 0; r < 4; ++r) {
        v16bf af = ldfrag(smem + XO + (r * 16 + lx) * DIMC + ks + kh * 8);
        acc[r] = wmma_bf16(af, bf, acc[r]);
      }
    }
    float bc = qkv_b[col0 + lx];
    int which = colTile / 32;                  // 0=q 1=k 2=v
    int cc = (colTile & 31) * 16 + lx;         // column within 512
    int h = cc >> 5, d = cc & 31;
#pragma unroll
    for (int r = 0; r < 4; ++r) {
#pragma unroll
      for (int e = 0; e < 8; ++e) {
        int n = r * 16 + e + kh * 8;           // C layout: M = e + 8*(lane>=16)
        float v = acc[r][e] + bc;
        if (which == 0)      smem[QS + (h * NPAD + n) * 32 + d] = f2bf(v * scale);
        else if (which == 1) smem[KS + (h * NPAD + n) * 32 + d] = f2bf(v);
        else                 smem[VT + (h * 32 + d) * NPAD + n] = f2bf(v);
      }
    }
  }
  __syncthreads();

  // ---- Phase 2: per-head attention (2 heads per wave) ----
  const int w = b & (NWIN - 1);
  for (int hh = 0; hh < 2; ++hh) {
    int h = wave * 2 + hh;
    for (int rt = 0; rt < 4; ++rt) {
      // S = q @ k^T  (head dim 32 == one bf16 WMMA K-step)
      v8f p[4] = {};
      v16bf aq = ldfrag(smem + QS + (h * NPAD + rt * 16 + lx) * 32 + kh * 8);
#pragma unroll
      for (int ct = 0; ct < 4; ++ct) {
        v16bf bk = ldfrag(smem + KS + (h * NPAD + ct * 16 + lx) * 32 + kh * 8);
        p[ct] = wmma_bf16(aq, bk, p[ct]);
      }
      // + relative-position bias + window mask; kill padded columns
#pragma unroll
      for (int ct = 0; ct < 4; ++ct) {
#pragma unroll
        for (int e = 0; e < 8; ++e) {
          int n = rt * 16 + e + kh * 8;
          int m = ct * 16 + lx;
          float v = p[ct][e];
          if (m >= NTOK) v = -1e30f;
          else if (n < NTOK)
            v += bias[(h * NTOK + n) * NTOK + m] +
                 mask[((size_t)w * NTOK + n) * NTOK + m];
          p[ct][e] = v;
        }
      }
      // row softmax: reduce across the 16-lane column group
      u16* pb = smem + PB + wave * 1024;   // per-wave 16x64 bf16 P tile
#pragma unroll
      for (int e = 0; e < 8; ++e) {
        float mx = fmaxf(fmaxf(p[0][e], p[1][e]), fmaxf(p[2][e], p[3][e]));
        for (int s = 1; s < 16; s <<= 1) mx = fmaxf(mx, __shfl_xor(mx, s, 32));
        float e0 = __expf(p[0][e] - mx), e1 = __expf(p[1][e] - mx);
        float e2 = __expf(p[2][e] - mx), e3 = __expf(p[3][e] - mx);
        float sm = e0 + e1 + e2 + e3;
        for (int s = 1; s < 16; s <<= 1) sm += __shfl_xor(sm, s, 32);
        float inv = 1.f / sm;
        int row = e + kh * 8;
        pb[row * 64 +  0 + lx] = f2bf(e0 * inv);
        pb[row * 64 + 16 + lx] = f2bf(e1 * inv);
        pb[row * 64 + 32 + lx] = f2bf(e2 * inv);
        pb[row * 64 + 48 + lx] = f2bf(e3 * inv);
      }
      // O = P @ V   (A = P from LDS, B = V^T rows [d][m])
      v8f o[2] = {};
#pragma unroll
      for (int ksm = 0; ksm < 2; ++ksm) {
        v16bf ap = ldfrag(pb + lx * 64 + ksm * 32 + kh * 8);
#pragma unroll
        for (int ct2 = 0; ct2 < 2; ++ct2) {
          v16bf bv = ldfrag(smem + VT + (h * 32 + ct2 * 16 + lx) * NPAD + ksm * 32 + kh * 8);
          o[ct2] = wmma_bf16(ap, bv, o[ct2]);
        }
      }
#pragma unroll
      for (int ct2 = 0; ct2 < 2; ++ct2) {
#pragma unroll
        for (int e = 0; e < 8; ++e) {
          int n = rt * 16 + e + kh * 8;
          int c = h * 32 + ct2 * 16 + lx;
          smem[XO + n * DIMC + c] = f2bf(o[ct2][e]);   // xs region reused as o
        }
      }
    }
  }
  __syncthreads();

  // ---- Phase 3: out = o @ W_proj + b ; store rows < 49 as fp32 ----
  for (int i = 0; i < 4; ++i) {
    int col0 = (wave * 4 + i) << 4;   // 32 column tiles over 512
    v8f acc[4] = {};
    for (int ks = 0; ks < DIMC; ks += 32) {
      v16bf bf = ldfrag(proj_wt + (size_t)(col0 + lx) * DIMC + ks + kh * 8);
#pragma unroll
      for (int r = 0; r < 4; ++r) {
        v16bf af = ldfrag(smem + XO + (r * 16 + lx) * DIMC + ks + kh * 8);
        acc[r] = wmma_bf16(af, bf, acc[r]);
      }
    }
    float bc = proj_b[col0 + lx];
#pragma unroll
    for (int r = 0; r < 4; ++r) {
#pragma unroll
      for (int e = 0; e < 8; ++e) {
        int n = r * 16 + e + kh * 8;
        if (n < NTOK)
          out[((size_t)b * NTOK + n) * DIMC + col0 + lx] = acc[r][e] + bc;
      }
    }
  }
}

extern "C" void kernel_launch(void* const* d_in, const int* in_sizes, int n_in,
                              void* d_out, int out_size, void* d_ws, size_t ws_size,
                              hipStream_t stream) {
  const float* x         = (const float*)d_in[0];
  const float* mask      = (const float*)d_in[1];
  const float* qkv_w     = (const float*)d_in[2];
  const float* qkv_b     = (const float*)d_in[3];
  const float* proj_w    = (const float*)d_in[4];
  const float* proj_b    = (const float*)d_in[5];
  const float* rel_table = (const float*)d_in[6];
  const int*   rel_index = (const int*)d_in[7];

  u16*   qkv_wt  = (u16*)d_ws;                                   // 1536*512 bf16
  u16*   proj_wt = qkv_wt + 1536 * 512;                          // 512*512 bf16
  float* biasp   = (float*)((char*)d_ws + (size_t)(1536 * 512 + 512 * 512) * 2);

  const int prep_total = 1536 * 512 + 512 * 512 + HEADS * NTOK * NTOK;
  prep_kernel<<<(prep_total + 255) / 256, 256, 0, stream>>>(
      qkv_w, proj_w, rel_table, rel_index, qkv_wt, proj_wt, biasp);

  attn_kernel<<<2048, 256, 0, stream>>>(
      x, mask, qkv_b, proj_b, qkv_wt, proj_wt, biasp, (float*)d_out);
}
